// MLPPredictor_64656437674425
// MI455X (gfx1250) — matrix operand order repaired and checked by reading, not compiled, not run on hardware
//
#include <hip/hip_runtime.h>
#include <hip/hip_bf16.h>

// MLP edge scorer for gfx1250 (MI455X).
// score[e] = 5*sigmoid( relu( [hu[src[e]], hi[dst[e]]] @ W1^T + b1 ) @ W2^T + b2 )
// Strategy: f16 WMMA 16x16x32 (f32 accumulate); W1 staged once per block in LDS as f16;
// each wave processes M=32 edges so every B-fragment (ds_load_b128 x2) feeds TWO wmmas;
// B fragments are software double-buffered so LDS latency overlaps the WMMA pipe.

typedef __attribute__((ext_vector_type(16))) _Float16 v16h;
typedef __attribute__((ext_vector_type(8)))  float    v8f;
typedef __attribute__((ext_vector_type(4)))  float    v4f;
typedef __attribute__((ext_vector_type(4)))  _Float16 v4h;

#define DFEAT      256
#define TWO_D      512
#define ROW_STRIDE 528                       // halves per W1 row in LDS (1056 B = 33*32, keeps 32B alignment)
#define LDS_BYTES  (DFEAT * ROW_STRIDE * 2)  // 270336 B < 320 KB WGP LDS
#define EDGES_PER_WAVE  32
#define EDGES_PER_BLOCK 256                  // 8 waves * 32 edges

extern __shared__ _Float16 w1s[];

__global__ __launch_bounds__(256)
void mlp_edge_score_kernel(const float* __restrict__ h_user,
                           const float* __restrict__ h_item,
                           const int*   __restrict__ src_idx,
                           const int*   __restrict__ dst_idx,
                           const float* __restrict__ W1,
                           const float* __restrict__ b1,
                           const float* __restrict__ W2,
                           const float* __restrict__ b2,
                           float* __restrict__ out,
                           int E, int nChunks)
{
    const int t = threadIdx.x;

    // ---- Stage W1 (f32 [256][512] row-major) into LDS as f16 with padded rows ----
    // 131072 floats / (256 threads * 4 per iter) = 128 iterations, coalesced.
    for (int i = 0; i < 128; ++i) {
        int f = (i * 256 + t) * 4;        // flat float index
        int n = f >> 9;                   // W1 row (output channel)
        int k = f & 511;                  // W1 col (input channel)
        v4f v = *(const v4f*)(W1 + f);
        v4h h;
        h[0] = (_Float16)v[0]; h[1] = (_Float16)v[1];
        h[2] = (_Float16)v[2]; h[3] = (_Float16)v[3];
        *(v4h*)(w1s + n * ROW_STRIDE + k) = h;
    }
    __syncthreads();

    const int lane = t & 31;
    const int wave = t >> 5;
    const int lh   = lane >> 4;      // lane half (0/1)
    const int j    = lane & 15;
    const float b2v = b2[0];

    for (int chunk = blockIdx.x; chunk < nChunks; chunk += gridDim.x) {
        const int tileBase = chunk * EDGES_PER_BLOCK + wave * EDGES_PER_WAVE;

        // ---- Gather x[e, 0:512] straight into WMMA A-fragment layout, 2 sub-tiles ----
        // A 16x32 f16: lane j holds M=j, K = kt*32 + lh*16 + 0..15
        v16h A[2][16];
#pragma unroll
        for (int sub = 0; sub < 2; ++sub) {
            int e = tileBase + sub * 16 + j;
            if (e >= E) e = E - 1;                      // clamp (E divisible by 256 in practice)
            const float* ub = h_user + (size_t)src_idx[e] * DFEAT;
            const float* ib = h_item + (size_t)dst_idx[e] * DFEAT;
#pragma unroll
            for (int kt = 0; kt < 16; ++kt) {
                const float* base = (kt < 8) ? (ub + kt * 32) : (ib + (kt - 8) * 32);
                const int f0 = lh * 16;
                v4f a = *(const v4f*)(base + f0);
                v4f b = *(const v4f*)(base + f0 + 4);
                v4f c = *(const v4f*)(base + f0 + 8);
                v4f d = *(const v4f*)(base + f0 + 12);
                v16h Af;
                Af[0]=(_Float16)a[0];  Af[1]=(_Float16)a[1];  Af[2]=(_Float16)a[2];  Af[3]=(_Float16)a[3];
                Af[4]=(_Float16)b[0];  Af[5]=(_Float16)b[1];  Af[6]=(_Float16)b[2];  Af[7]=(_Float16)b[3];
                Af[8]=(_Float16)c[0];  Af[9]=(_Float16)c[1];  Af[10]=(_Float16)c[2]; Af[11]=(_Float16)c[3];
                Af[12]=(_Float16)d[0]; Af[13]=(_Float16)d[1]; Af[14]=(_Float16)d[2]; Af[15]=(_Float16)d[3];
                A[sub][kt] = Af;
            }
        }

        float s[2][8];
#pragma unroll
        for (int r = 0; r < 8; ++r) { s[0][r] = 0.f; s[1][r] = 0.f; }

        // ---- 16 n-tiles x 16 k-tiles; each B fragment feeds 2 wmmas (M=32) ----
        for (int nt = 0; nt < 16; ++nt) {
            const int n = nt * 16 + j;                      // output channel handled by this lane
            const _Float16* brow = w1s + n * ROW_STRIDE + lh * 16;
            v8f acc0 = {};
            v8f acc1 = {};
            // Software double-buffer: keep next B-fragment's ds_loads in flight
            // while the current one feeds the WMMA pipe.
            v16h Bcur = *(const v16h*)(brow);
#pragma unroll
            for (int kt = 0; kt < 16; ++kt) {
                v16h Bnext;
                if (kt < 15) Bnext = *(const v16h*)(brow + (kt + 1) * 32);
                // B 32x16 f16: lane (lh,j) holds column n=j, K = kt*32 + lh*16 + 0..15
                acc0 = __builtin_amdgcn_wmma_f32_16x16x32_f16(
                           false, A[0][kt], false, Bcur, (short)0, acc0, false, false);
                acc1 = __builtin_amdgcn_wmma_f32_16x16x32_f16(
                           false, A[1][kt], false, Bcur, (short)0, acc1, false, false);
                if (kt < 15) Bcur = Bnext;
            }
            // Fused bias + ReLU + W2 partial dot.
            // C layout: acc[r] = h[m = r + 8*lh, n = nt*16 + j]
            const float b1v = b1[n];
            const float w2v = W2[n];
#pragma unroll
            for (int r = 0; r < 8; ++r) {
                float h0 = acc0[r] + b1v;
                float h1 = acc1[r] + b1v;
                h0 = h0 > 0.f ? h0 : 0.f;
                h1 = h1 > 0.f ? h1 : 0.f;
                s[0][r] += h0 * w2v;
                s[1][r] += h1 * w2v;
            }
        }

        // ---- Reduce partial dots across the 16 lanes of each half ----
#pragma unroll
        for (int sub = 0; sub < 2; ++sub) {
#pragma unroll
            for (int r = 0; r < 8; ++r) {
                s[sub][r] += __shfl_xor(s[sub][r], 1, 32);
                s[sub][r] += __shfl_xor(s[sub][r], 2, 32);
                s[sub][r] += __shfl_xor(s[sub][r], 4, 32);
                s[sub][r] += __shfl_xor(s[sub][r], 8, 32);
            }
        }

        // Edge m = 8*lh + r within each sub-tile; lanes j<8 of each half write one score each.
        if (j < 8) {
#pragma unroll
            for (int sub = 0; sub < 2; ++sub) {
                const int eo = tileBase + sub * 16 + lh * 8 + j;
                if (eo < E) {
                    const float x = s[sub][j] + b2v;
                    out[eo] = 5.0f / (1.0f + __expf(-x));
                }
            }
        }
    }
}

extern "C" void kernel_launch(void* const* d_in, const int* in_sizes, int n_in,
                              void* d_out, int out_size, void* d_ws, size_t ws_size,
                              hipStream_t stream) {
    const float* h_user  = (const float*)d_in[0];
    const float* h_item  = (const float*)d_in[1];
    const int*   src_idx = (const int*)d_in[2];
    const int*   dst_idx = (const int*)d_in[3];
    const float* W1      = (const float*)d_in[4];
    const float* b1      = (const float*)d_in[5];
    const float* W2      = (const float*)d_in[6];
    const float* b2      = (const float*)d_in[7];
    float*       out     = (float*)d_out;

    const int E = in_sizes[2];                                  // number of edges
    const int nChunks = (E + EDGES_PER_BLOCK - 1) / EDGES_PER_BLOCK;
    int blocks = nChunks < 240 ? nChunks : 240;

    mlp_edge_score_kernel<<<blocks, 256, LDS_BYTES, stream>>>(
        h_user, h_item, src_idx, dst_idx, W1, b1, W2, b2, out, E, nChunks);
}
